// TransformerBlock_46471546142871
// MI455X (gfx1250) — matrix operand order repaired and checked by reading, not compiled, not run on hardware
//
#include <hip/hip_runtime.h>

// ---------------- problem constants ----------------
#define B_  4
#define S_  2048
#define D_  1024
#define H_  16
#define HD_ 64
#define FF_ 4096
#define M_  (B_ * S_)   // 8192 rows of activations

// ---------------- vector types for WMMA ----------------
typedef __attribute__((ext_vector_type(16))) __bf16 v16bf;
typedef __attribute__((ext_vector_type(8)))  __bf16 v8bf;
typedef __attribute__((ext_vector_type(8)))  float  v8f;

typedef unsigned short u16;   // bf16 storage type (host-safe)

// ---------------- helpers ----------------
__device__ __forceinline__ u16 f2bf(float f) {
  union { float f; unsigned u; } v; v.f = f;
  unsigned r = v.u + 0x7FFFu + ((v.u >> 16) & 1u);   // round-to-nearest-even
  return (u16)(r >> 16);
}

// A-matrix fragment (16x32 bf16): lane L holds row (L&15); K-chunks
// {0..15} for lanes 0-15, {16..31} for lanes 16-31.
__device__ __forceinline__ v16bf load_a_frag(const u16* p0, int ld, int lane) {
  const u16* p = p0 + (size_t)(lane & 15) * ld + ((lane >> 4) << 4);
  v8bf lo = *(const v8bf*)p;          // K = base .. base+7
  v8bf hi = *(const v8bf*)(p + 8);    // K = base+8 .. base+15
  return __builtin_shufflevector(lo, hi, 0,1,2,3,4,5,6,7,8,9,10,11,12,13,14,15);
}

// B-matrix fragment (32x16 bf16) from a transposed (N-major, K contiguous)
// matrix: lane n<16 reads Wt[n][k..k+15], lanes 16-31 read [k+16..k+31].
__device__ __forceinline__ v16bf load_b_frag(const u16* p0, int ld, int lane) {
  const u16* p = p0 + (size_t)(lane & 15) * ld + ((lane >> 4) << 4);
  return *(const v16bf*)p;
}

__device__ __forceinline__ v8f wmma_bf16(v16bf a, v16bf b, v8f c) {
  return __builtin_amdgcn_wmma_f32_16x16x32_bf16(false, a, false, b, (short)0, c,
                                                 false, false);
}

__device__ __forceinline__ float gelu_tanh(float v) {
  float u = 0.7978845608028654f * (v + 0.044715f * v * v * v);
  return 0.5f * v * (1.0f + tanhf(u));
}

// ---------------- weight transpose: W[K,N] f32 -> Wt[N,K] bf16 ----------------
__global__ __launch_bounds__(256) void k_transpose_w(const float* __restrict__ W,
                                                     u16* __restrict__ Wt,
                                                     int K, int N) {
  __shared__ float tile[32][33];
  int tx = threadIdx.x & 31, ty = threadIdx.x >> 5;
  int n0 = blockIdx.x * 32, k0 = blockIdx.y * 32;
#pragma unroll
  for (int r = 0; r < 4; ++r) {
    int k = k0 + ty + r * 8;
    tile[ty + r * 8][tx] = W[(size_t)k * N + n0 + tx];
  }
  __syncthreads();
#pragma unroll
  for (int r = 0; r < 4; ++r) {
    int n = n0 + ty + r * 8;
    Wt[(size_t)n * K + k0 + tx] = f2bf(tile[tx][ty + r * 8]);
  }
}

// ---------------- V transpose (per batch): V[S,D] bf16 -> Vt[D,S] bf16 -------
__global__ __launch_bounds__(256) void k_transpose_v(const u16* __restrict__ V,
                                                     u16* __restrict__ Vt) {
  __shared__ u16 tile[32][33];
  int tx = threadIdx.x & 31, ty = threadIdx.x >> 5;
  int b = blockIdx.z;
  int d0 = blockIdx.x * 32, s0 = blockIdx.y * 32;
#pragma unroll
  for (int r = 0; r < 4; ++r) {
    int s = s0 + ty + r * 8;
    tile[ty + r * 8][tx] = V[((size_t)(b * S_ + s)) * D_ + d0 + tx];
  }
  __syncthreads();
#pragma unroll
  for (int r = 0; r < 4; ++r) {
    int d = d0 + ty + r * 8;
    Vt[((size_t)(b * D_ + d)) * S_ + s0 + tx] = tile[tx][ty + r * 8];
  }
}

// ---------------- layernorm (f32 in) -> bf16 out ----------------
__global__ __launch_bounds__(256) void k_layernorm(const float* __restrict__ x,
                                                   const float* __restrict__ g,
                                                   const float* __restrict__ be,
                                                   u16* __restrict__ out) {
  int row = blockIdx.x;
  const float* xr = x + (size_t)row * D_;
  int t = threadIdx.x;
  float v[4], s = 0.f, ss = 0.f;
#pragma unroll
  for (int i = 0; i < 4; ++i) {
    float f = xr[t + i * 256];
    v[i] = f; s += f; ss += f * f;
  }
#pragma unroll
  for (int off = 16; off >= 1; off >>= 1) {
    s  += __shfl_xor(s,  off);
    ss += __shfl_xor(ss, off);
  }
  __shared__ float sm[8], sm2[8];
  int wv = t >> 5, lane = t & 31;
  if (lane == 0) { sm[wv] = s; sm2[wv] = ss; }
  __syncthreads();
  s = 0.f; ss = 0.f;
#pragma unroll
  for (int i = 0; i < 8; ++i) { s += sm[i]; ss += sm2[i]; }
  float mean = s * (1.0f / D_);
  float var  = ss * (1.0f / D_) - mean * mean;
  float rstd = rsqrtf(var + 1e-5f);
#pragma unroll
  for (int i = 0; i < 4; ++i) {
    int c = t + i * 256;
    out[(size_t)row * D_ + c] = f2bf(g[c] * (v[i] - mean) * rstd + be[c]);
  }
}

// ---------------- WMMA GEMM with async-LDS weight staging --------------------
// C[M,N] = act(scale * A @ Wt^T + bias) + resid
// A: [M,K] bf16 row-major.  Wt: [N,K] bf16 row-major (pre-transposed weight).
// Block: 256 threads = 8 waves; block tile 256(M) x 64(N); wave tile 32x64.
// The 64x32 weight tile (4KB) shared by all waves is staged in LDS with
// gfx1250 global_load_async_to_lds_b128 (ASYNCcnt), double-buffered.
// LDS column stride = 40 elements (80B): 20-bank shift per column gives a
// conflict-free permutation of all 64 banks for the b128 fragment reads,
// while keeping 16B alignment.
#define BSTRIDE 40

__global__ __launch_bounds__(256) void k_gemm_wmma(const u16* __restrict__ A,
                                                   const u16* __restrict__ Wt,
                                                   float* __restrict__ outF,
                                                   u16* __restrict__ outH,
                                                   const float* __restrict__ bias,
                                                   const float* __restrict__ resid,
                                                   int M, int N, int K,
                                                   float scale, int act) {
  __shared__ u16 bstage[2][64 * BSTRIDE];   // 2 x 5120B

  int tid  = threadIdx.x;
  int lane = tid & 31, wv = tid >> 5;
  int m0 = blockIdx.y * 256 + wv * 32;
  int n0 = blockIdx.x * 64;

  // async-loader assignment: thread -> (column 0..63, 8-element k-chunk)
  int lcol = tid >> 2;            // 0..63
  int lk8  = (tid & 3) << 3;      // 0,8,16,24
  const u16* wsrc = Wt + (size_t)(n0 + lcol) * K + lk8;
  unsigned loff0 = (unsigned)(unsigned long long)&bstage[0][lcol * BSTRIDE + lk8];
  unsigned loff1 = (unsigned)(unsigned long long)&bstage[1][lcol * BSTRIDE + lk8];

  const u16* abase0 = A + (size_t)m0 * K;
  const u16* abase1 = A + (size_t)(m0 + 16) * K;

  v8f acc[2][4] = {};
  int nk = K >> 5;

  // prologue: stage tile 0 into buffer 0
  {
    const u16* src = wsrc;
    asm volatile("global_load_async_to_lds_b128 %0, %1, off"
                 :: "v"(loff0), "v"(src) : "memory");
  }

#pragma unroll 1
  for (int ki = 0; ki < nk; ++ki) {
    int cur = ki & 1;
    int k = ki << 5;
    // stage next tile into the other buffer (wrap on last iter; uniform wait)
    {
      int kn = (ki + 1 < nk) ? (k + 32) : 0;
      const u16* src = wsrc + kn;
      unsigned loff = cur ? loff0 : loff1;
      asm volatile("global_load_async_to_lds_b128 %0, %1, off"
                   :: "v"(loff), "v"(src) : "memory");
    }
    // wait for our tile(ki) issue (only tile(ki+1) may remain outstanding)
    asm volatile("s_wait_asynccnt 0x1" ::: "memory");
    __syncthreads();   // tile(ki) visible to all waves

    // prefetch the A stream two tiles ahead
    if (ki + 2 < nk) {
      const u16* pf = abase0 + (size_t)(lane & 15) * K + (k + 64) + ((lane >> 4) << 3);
      __builtin_prefetch(pf, 0, 3);
      __builtin_prefetch(pf + (size_t)16 * K, 0, 3);
    }

    // issue all fragment loads first, then run 8 WMMAs back-to-back
    v16bf a0 = load_a_frag(abase0 + k, K, lane);
    v16bf a1 = load_a_frag(abase1 + k, K, lane);
    v16bf bf[4];
#pragma unroll
    for (int t = 0; t < 4; ++t) {
      const u16* bp = &bstage[cur][(t * 16 + (lane & 15)) * BSTRIDE +
                                   ((lane >> 4) << 4)];
      bf[t] = *(const v16bf*)bp;
    }
#pragma unroll
    for (int t = 0; t < 4; ++t) {
      acc[0][t] = wmma_bf16(a0, bf[t], acc[0][t]);
      acc[1][t] = wmma_bf16(a1, bf[t], acc[1][t]);
    }
    __syncthreads();   // everyone done reading buf[cur] before it is re-staged
  }

  // ---- epilogue ----
#pragma unroll
  for (int r = 0; r < 2; ++r) {
    int rbase = m0 + r * 16 + ((lane >> 4) << 3);
#pragma unroll
    for (int t = 0; t < 4; ++t) {
      int n = n0 + t * 16 + (lane & 15);
      float bv = bias ? bias[n] : 0.0f;
#pragma unroll
      for (int i = 0; i < 8; ++i) {
        float v = acc[r][t][i] * scale + bv;
        if (act) v = gelu_tanh(v);
        size_t idx = (size_t)(rbase + i) * N + n;
        if (resid) v += resid[idx];
        if (outF) outF[idx] = v;
        if (outH) outH[idx] = f2bf(v);
      }
    }
  }
}

// ---------------- flash attention (causal, per-head) ----------------
// Q,K: [B,S,D] bf16 (Q pre-scaled by 1/sqrt(HD)).  Vt: [B, H*64, S] bf16.
// One wave per 16-query tile; 8 waves per block.
__global__ __launch_bounds__(256) void k_flash_attn(const u16* __restrict__ Q,
                                                    const u16* __restrict__ Km,
                                                    const u16* __restrict__ Vt,
                                                    u16* __restrict__ ctx) {
  __shared__ u16 plds[8][16 * 32];   // per-wave P staging (C-layout -> A-layout)
  int lane = threadIdx.x & 31, wv = threadIdx.x >> 5;
  int bh = blockIdx.y;
  int b = bh >> 4, h = bh & 15;
  int q0 = (blockIdx.x * 8 + wv) * 16;

  const u16* qbase = Q + ((size_t)(b * S_ + q0)) * D_ + h * HD_;
  v16bf qa0 = load_a_frag(qbase,      D_, lane);   // head-dim 0..31
  v16bf qa1 = load_a_frag(qbase + 32, D_, lane);   // head-dim 32..63

  float m[8], l[8];
  v8f c[4] = {};
#pragma unroll
  for (int i = 0; i < 8; ++i) { m[i] = -1e30f; l[i] = 0.0f; }
  u16* myp = plds[wv];

  for (int j0 = 0; j0 <= q0; j0 += 32) {
    // ---- scores: two 16x16 tiles (keys j0..j0+15 and j0+16..j0+31) ----
    // load all 4 K fragments first, then 4 WMMAs back-to-back
    v16bf kf[2][2];
#pragma unroll
    for (int t = 0; t < 2; ++t) {
      const u16* kb = Km + ((size_t)(b * S_ + j0 + t * 16)) * D_ + h * HD_;
      kf[t][0] = load_b_frag(kb,      D_, lane);
      kf[t][1] = load_b_frag(kb + 32, D_, lane);
    }
    v8f s[2];
#pragma unroll
    for (int t = 0; t < 2; ++t) {
      v8f sa = {};
      sa = wmma_bf16(qa0, kf[t][0], sa);
      sa = wmma_bf16(qa1, kf[t][1], sa);
      s[t] = sa;
    }
    // ---- causal mask + online softmax (rows split across 16-lane halves) ----
    float alpha[8];
#pragma unroll
    for (int i = 0; i < 8; ++i) {
      int rowg = q0 + i + ((lane >> 4) << 3);
      int key0 = j0 + (lane & 15);
      float s0 = (key0      <= rowg) ? s[0][i] : -1e30f;
      float s1 = (key0 + 16 <= rowg) ? s[1][i] : -1e30f;
      float rmax = fmaxf(s0, s1);
#pragma unroll
      for (int off = 8; off >= 1; off >>= 1)
        rmax = fmaxf(rmax, __shfl_xor(rmax, off));
      float mn = fmaxf(m[i], rmax);
      alpha[i] = __expf(m[i] - mn);
      float p0 = __expf(s0 - mn);
      float p1 = __expf(s1 - mn);
      float rs = p0 + p1;
#pragma unroll
      for (int off = 8; off >= 1; off >>= 1)
        rs += __shfl_xor(rs, off);
      l[i] = l[i] * alpha[i] + rs;
      m[i] = mn;
      int prow = i + ((lane >> 4) << 3);
      myp[prow * 32 + (lane & 15)]      = f2bf(p0);
      myp[prow * 32 + (lane & 15) + 16] = f2bf(p1);
    }
    // rescale running context
#pragma unroll
    for (int t = 0; t < 4; ++t)
#pragma unroll
      for (int i = 0; i < 8; ++i) c[t][i] *= alpha[i];

    // ---- ctx += P @ V  (P from LDS in A-layout, Vt rows contiguous) ----
    {
      int r = lane & 15, c0 = (lane >> 4) << 4;
      v8bf lo = *(const v8bf*)&myp[r * 32 + c0];
      v8bf hi = *(const v8bf*)&myp[r * 32 + c0 + 8];
      v16bf pa = __builtin_shufflevector(lo, hi,
                   0,1,2,3,4,5,6,7,8,9,10,11,12,13,14,15);
      v16bf vb[4];
#pragma unroll
      for (int t = 0; t < 4; ++t) {
        const u16* vp = Vt + ((size_t)(b * D_ + h * HD_ + t * 16)) * S_ + j0;
        vb[t] = load_b_frag(vp, S_, lane);
      }
#pragma unroll
      for (int t = 0; t < 4; ++t)
        c[t] = wmma_bf16(pa, vb[t], c[t]);
    }
  }

  // ---- normalize and store ctx (bf16) ----
  float inv[8];
#pragma unroll
  for (int i = 0; i < 8; ++i) inv[i] = 1.0f / l[i];
#pragma unroll
  for (int t = 0; t < 4; ++t) {
    int col = h * HD_ + t * 16 + (lane & 15);
#pragma unroll
    for (int i = 0; i < 8; ++i) {
      int rowg = q0 + i + ((lane >> 4) << 3);
      ctx[((size_t)(b * S_ + rowg)) * D_ + col] = f2bf(c[t][i] * inv[i]);
    }
  }
}

// ---------------- host orchestration ----------------
extern "C" void kernel_launch(void* const* d_in, const int* in_sizes, int n_in,
                              void* d_out, int out_size, void* d_ws, size_t ws_size,
                              hipStream_t stream) {
  const float* x   = (const float*)d_in[0];
  const float* Wq  = (const float*)d_in[1];
  const float* Wk  = (const float*)d_in[2];
  const float* Wv  = (const float*)d_in[3];
  const float* Wo  = (const float*)d_in[4];
  const float* bo  = (const float*)d_in[5];
  const float* W1  = (const float*)d_in[6];
  const float* b1  = (const float*)d_in[7];
  const float* W2  = (const float*)d_in[8];
  const float* b2  = (const float*)d_in[9];
  const float* g1  = (const float*)d_in[10];
  const float* be1 = (const float*)d_in[11];
  const float* g2  = (const float*)d_in[12];
  const float* be2 = (const float*)d_in[13];
  float* out = (float*)d_out;

  // workspace layout (bf16 buffers as u16), 256B aligned
  char* ws = (char*)d_ws;
  size_t off = 0;
  auto alloc = [&](size_t bytes) -> void* {
    void* p = ws + off;
    off += (bytes + 255) & ~(size_t)255;
    return p;
  };
  u16* WqT = (u16*)alloc((size_t)D_ * D_ * 2);
  u16* WkT = (u16*)alloc((size_t)D_ * D_ * 2);
  u16* WvT = (u16*)alloc((size_t)D_ * D_ * 2);
  u16* WoT = (u16*)alloc((size_t)D_ * D_ * 2);
  u16* W1T = (u16*)alloc((size_t)D_ * FF_ * 2);
  u16* W2T = (u16*)alloc((size_t)D_ * FF_ * 2);
  u16* hB  = (u16*)alloc((size_t)M_ * D_ * 2);   // LN1 out; reused as ctx
  u16* Qb  = (u16*)alloc((size_t)M_ * D_ * 2);
  u16* Kb  = (u16*)alloc((size_t)M_ * D_ * 2);
  u16* Vb  = (u16*)alloc((size_t)M_ * D_ * 2);   // reused as LN2 out
  u16* Vt  = (u16*)alloc((size_t)M_ * D_ * 2);
  float* x1 = (float*)alloc((size_t)M_ * D_ * 4);
  u16* ffB = (u16*)alloc((size_t)M_ * FF_ * 2);
  u16* ctx = hB;   // h dead after QKV GEMMs
  u16* h2B = Vb;   // V dead after transpose into Vt

  dim3 tb(256);
  // 1) weight transposes (f32 -> bf16, N-major)
  k_transpose_w<<<dim3(D_ / 32, D_ / 32), tb, 0, stream>>>(Wq, WqT, D_, D_);
  k_transpose_w<<<dim3(D_ / 32, D_ / 32), tb, 0, stream>>>(Wk, WkT, D_, D_);
  k_transpose_w<<<dim3(D_ / 32, D_ / 32), tb, 0, stream>>>(Wv, WvT, D_, D_);
  k_transpose_w<<<dim3(D_ / 32, D_ / 32), tb, 0, stream>>>(Wo, WoT, D_, D_);
  k_transpose_w<<<dim3(FF_ / 32, D_ / 32), tb, 0, stream>>>(W1, W1T, D_, FF_);
  k_transpose_w<<<dim3(D_ / 32, FF_ / 32), tb, 0, stream>>>(W2, W2T, FF_, D_);

  // 2) LN1 -> bf16
  k_layernorm<<<M_, tb, 0, stream>>>(x, g1, be1, hB);

  // 3) QKV projections (Q pre-scaled by 1/sqrt(HD) = 0.125)
  dim3 gD(D_ / 64, M_ / 256);
  k_gemm_wmma<<<gD, tb, 0, stream>>>(hB, WqT, nullptr, Qb, nullptr, nullptr,
                                     M_, D_, D_, 0.125f, 0);
  k_gemm_wmma<<<gD, tb, 0, stream>>>(hB, WkT, nullptr, Kb, nullptr, nullptr,
                                     M_, D_, D_, 1.0f, 0);
  k_gemm_wmma<<<gD, tb, 0, stream>>>(hB, WvT, nullptr, Vb, nullptr, nullptr,
                                     M_, D_, D_, 1.0f, 0);

  // 4) V -> Vt (per batch [S,D] -> [D,S]) so ctx WMMA B-frags are contiguous
  k_transpose_v<<<dim3(D_ / 32, S_ / 32, B_), tb, 0, stream>>>(Vb, Vt);

  // 5) causal flash attention -> ctx (bf16)
  k_flash_attn<<<dim3(S_ / 128, B_ * H_), tb, 0, stream>>>(Qb, Kb, Vt, ctx);

  // 6) output projection + bias + residual -> x1 (f32)
  k_gemm_wmma<<<gD, tb, 0, stream>>>(ctx, WoT, x1, nullptr, bo, x,
                                     M_, D_, D_, 1.0f, 0);

  // 7) LN2 -> bf16
  k_layernorm<<<M_, tb, 0, stream>>>(x1, g2, be2, h2B);

  // 8) FF1 + bias + GELU -> bf16
  k_gemm_wmma<<<dim3(FF_ / 64, M_ / 256), tb, 0, stream>>>(
      h2B, W1T, nullptr, ffB, b1, nullptr, M_, FF_, D_, 1.0f, 1);

  // 9) FF2 + bias + residual -> d_out (f32)
  k_gemm_wmma<<<gD, tb, 0, stream>>>(ffB, W2T, out, nullptr, b2, x1,
                                     M_, D_, FF_, 1.0f, 0);
}
